// ViewEncoder_72834055406012
// MI455X (gfx1250) — matrix operand order repaired and checked by reading, not compiled, not run on hardware
//
#include <hip/hip_runtime.h>
#include <hip/hip_bf16.h>
#include <math.h>

// ---------- WMMA types (CDNA5 / gfx1250, wave32) ----------
typedef __attribute__((ext_vector_type(16))) __bf16 v16bf;
typedef __attribute__((ext_vector_type(8)))  float  v8f;

#define H 128

// Load + convert the per-lane A fragment (16x32 bf16 tile) for one WMMA k-step.
// ISA layout: lanes 0-15 row M=lane&15, K = {kbase+0..7, kbase+16..23};
// lanes 16-31 same rows, K offset +8 (kbase includes that).
// Hardware cvt: pairs lower to v_cvt_pk_bf16_f32.
__device__ __forceinline__ v16bf load_a_frag(const float* __restrict__ rowPtr, int kbase) {
    const float4* p0 = (const float4*)(rowPtr + kbase);
    const float4* p1 = (const float4*)(rowPtr + kbase + 16);
    float4 q0 = p0[0], q1 = p0[1], q2 = p1[0], q3 = p1[1];
    v16bf a;
    a[0]  = (__bf16)q0.x; a[1]  = (__bf16)q0.y; a[2]  = (__bf16)q0.z; a[3]  = (__bf16)q0.w;
    a[4]  = (__bf16)q1.x; a[5]  = (__bf16)q1.y; a[6]  = (__bf16)q1.z; a[7]  = (__bf16)q1.w;
    a[8]  = (__bf16)q2.x; a[9]  = (__bf16)q2.y; a[10] = (__bf16)q2.z; a[11] = (__bf16)q2.w;
    a[12] = (__bf16)q3.x; a[13] = (__bf16)q3.y; a[14] = (__bf16)q3.z; a[15] = (__bf16)q3.w;
    return a;
}

// ---------------------------------------------------------------------------
// Weight pre-packing into per-lane WMMA B-fragment order (bf16).
// Fragment element e of lane L at (colTile ct, kstep kk):
//   col = ct*16 + (L&15),  k = kk*32 + (L>=16 ? 8 : 0) + (e&7) + (e>=8 ? 16 : 0)
// Packed linear index: ((ct*4 + kk)*32 + L)*16 + e  -> each lane reads one
// contiguous 32-byte v16bf (two global_load_b128).
// ---------------------------------------------------------------------------

// B(k,c) = W[l][k*H + c]  (msg GEMM uses W directly), 8 col tiles per layer.
__global__ __launch_bounds__(256) void gnn_pack_W(const float* __restrict__ W,
                                                  __bf16* __restrict__ out, int layers) {
    int idx = blockIdx.x * 256 + threadIdx.x;
    int total = layers * 8 * 4 * 32 * 16;
    if (idx >= total) return;
    int e    = idx & 15;
    int lane = (idx >> 4) & 31;
    int kk   = (idx >> 9) & 3;
    int ct   = (idx >> 11) & 7;
    int l    = idx >> 14;
    int col = ct * 16 + (lane & 15);
    int k   = kk * 32 + ((lane >> 4) * 8) + (e & 7) + ((e >> 3) * 16);
    out[idx] = (__bf16)W[(size_t)l * H * H + (size_t)k * H + col];
}

// B(k,c) = w[c*H + k]  (GRU uses w^T), 24 col tiles (3H=384 outputs).
__global__ __launch_bounds__(256) void gnn_pack_wT(const float* __restrict__ w,
                                                   __bf16* __restrict__ out) {
    int idx = blockIdx.x * 256 + threadIdx.x;
    const int total = 24 * 4 * 32 * 16;
    if (idx >= total) return;
    int e    = idx & 15;
    int lane = (idx >> 4) & 31;
    int kk   = (idx >> 9) & 3;
    int ct   = idx >> 11;
    int col = ct * 16 + (lane & 15);
    int k   = kk * 32 + ((lane >> 4) * 8) + (e & 7) + ((e >> 3) * 16);
    out[idx] = (__bf16)w[(size_t)col * H + k];
}

// ---------------------------------------------------------------------------
// Kernel 1: m = h @ Wl   (N x 128) @ (128 x 128)
// Block: 256 threads = 8 waves, 16 rows x 128 cols. Wave w owns col tile 16w.
// ---------------------------------------------------------------------------
__global__ __launch_bounds__(256) void gnn_gemm_msg(const float* __restrict__ h,
                                                    const v16bf* __restrict__ Wp,
                                                    float* __restrict__ m, int N) {
    const int rowBase = blockIdx.x * 16;
    if (rowBase >= N) return;
    const int wave = threadIdx.x >> 5;       // = column tile ct
    const int lane = threadIdx.x & 31;
    const int mrow = rowBase + (lane & 15);
    const int hiA = (lane >> 4) * 8;
    const int col = wave * 16 + (lane & 15);

    const float* rowPtr = h + (size_t)mrow * H;
    v8f acc = {};
#pragma unroll
    for (int kk = 0; kk < 4; ++kk) {
        v16bf a = load_a_frag(rowPtr, kk * 32 + hiA);
        v16bf b = Wp[((size_t)wave * 4 + kk) * 32 + lane];
        acc = __builtin_amdgcn_wmma_f32_16x16x32_bf16(false, a, false, b,
                                                      (short)0, acc, false, false);
    }
    const int rOff = rowBase + (lane >> 4) * 8;
#pragma unroll
    for (int j = 0; j < 8; ++j) {
        m[(size_t)(rOff + j) * H + col] = acc[j];
    }
}

// ---------------------------------------------------------------------------
// Kernel 2: zero agg (float4 stores)
// ---------------------------------------------------------------------------
__global__ __launch_bounds__(256) void gnn_zero4(float4* __restrict__ p, long long n4) {
    long long i = (long long)blockIdx.x * 256 + threadIdx.x;
    if (i < n4) p[i] = make_float4(0.f, 0.f, 0.f, 0.f);
}

// ---------------------------------------------------------------------------
// Kernel 3: scatter-add  agg[dst] += m[src]  over E edges.
// One wave per edge; lane handles 4 floats (float4 load, 4 f32 atomics -> L2).
// ---------------------------------------------------------------------------
__global__ __launch_bounds__(256) void gnn_scatter(const float* __restrict__ m,
                                                   const int* __restrict__ src,
                                                   const int* __restrict__ dst,
                                                   float* __restrict__ agg, long long E) {
    long long e = (long long)blockIdx.x * 8 + (threadIdx.x >> 5);
    if (e >= E) return;
    const int lane = threadIdx.x & 31;
    __builtin_prefetch(src + e + 4096, 0, 1);   // global_prefetch on index stream
    __builtin_prefetch(dst + e + 4096, 0, 1);
    const int s = src[e];
    const int d = dst[e];
    float4 v = *((const float4*)(m + (size_t)s * H) + lane);
    float* ap = agg + (size_t)d * H + lane * 4;
    atomicAdd(ap + 0, v.x);
    atomicAdd(ap + 1, v.y);
    atomicAdd(ap + 2, v.z);
    atomicAdd(ap + 3, v.w);
}

// ---------------------------------------------------------------------------
// Kernel 4: fused GRU cell.
// Block = 256 threads = 8 waves, 16 rows. 48 output tiles (24 gi + 24 gh);
// wave w computes tiles [6w, 6w+6) -> LDS, then fused gates, in-place update.
// ---------------------------------------------------------------------------
__global__ __launch_bounds__(256) void gnn_gru(const float* __restrict__ agg,
                                               const float* __restrict__ hin,
                                               const v16bf* __restrict__ wihP,
                                               const v16bf* __restrict__ whhP,
                                               const float* __restrict__ b_ih,
                                               const float* __restrict__ b_hh,
                                               float* __restrict__ hout, int N) {
    __shared__ float gi[16][392];   // 384 + pad
    __shared__ float gh[16][392];

    const int rowBase = blockIdx.x * 16;
    if (rowBase >= N) return;
    const int wave = threadIdx.x >> 5;
    const int lane = threadIdx.x & 31;
    const int mrow = rowBase + (lane & 15);
    const int hiA = (lane >> 4) * 8;

    // Preload A fragments for all 4 k-steps, for both agg and h operands.
    v16bf Aa[4], Ah[4];
    const float* aRow = agg + (size_t)mrow * H;
    const float* hRow = hin + (size_t)mrow * H;
#pragma unroll
    for (int kk = 0; kk < 4; ++kk) {
        Aa[kk] = load_a_frag(aRow, kk * 32 + hiA);
        Ah[kk] = load_a_frag(hRow, kk * 32 + hiA);
    }

#pragma unroll
    for (int tt = 0; tt < 6; ++tt) {
        const int t = wave * 6 + tt;           // 0..47
        const bool isGi = (t < 24);
        const int ct = isGi ? t : t - 24;      // column tile within [0,384)
        const v16bf* Bp = isGi ? wihP : whhP;

        v8f acc = {};
#pragma unroll
        for (int kk = 0; kk < 4; ++kk) {
            v16bf b = Bp[((size_t)ct * 4 + kk) * 32 + lane];
            v16bf a = isGi ? Aa[kk] : Ah[kk];
            acc = __builtin_amdgcn_wmma_f32_16x16x32_bf16(false, a, false, b,
                                                          (short)0, acc, false, false);
        }
        const int rloc = (lane >> 4) * 8;
        const int cloc = ct * 16 + (lane & 15);
        if (isGi) {
#pragma unroll
            for (int j = 0; j < 8; ++j) gi[rloc + j][cloc] = acc[j];
        } else {
#pragma unroll
            for (int j = 0; j < 8; ++j) gh[rloc + j][cloc] = acc[j];
        }
    }
    __syncthreads();

    // Fused gates: 16 rows x 128 cols = 2048 elements, 8 per thread.
#pragma unroll
    for (int i = threadIdx.x; i < 16 * H; i += 256) {
        const int r = i >> 7;
        const int c = i & (H - 1);
        float ir  = gi[r][c]       + b_ih[c];
        float iz  = gi[r][c + 128] + b_ih[c + 128];
        float inn = gi[r][c + 256] + b_ih[c + 256];
        float hr  = gh[r][c]       + b_hh[c];
        float hz  = gh[r][c + 128] + b_hh[c + 128];
        float hn  = gh[r][c + 256] + b_hh[c + 256];
        float rg = 1.f / (1.f + __expf(-(ir + hr)));
        float zg = 1.f / (1.f + __expf(-(iz + hz)));
        float ng = tanhf(inn + rg * hn);
        float hold = hin[(size_t)(rowBase + r) * H + c];
        hout[(size_t)(rowBase + r) * H + c] = (1.f - zg) * ng + zg * hold;
    }
}

// ---------------------------------------------------------------------------
// Kernel 5: LayerNorm + residual.  One wave per row (wave32 butterfly reduce).
// ---------------------------------------------------------------------------
__global__ __launch_bounds__(256) void gnn_layernorm(const float* __restrict__ h,
                                                     const float* __restrict__ x,
                                                     const float* __restrict__ gamma,
                                                     const float* __restrict__ beta,
                                                     float* __restrict__ out, int N) {
    const int row = blockIdx.x * 8 + (threadIdx.x >> 5);
    if (row >= N) return;
    const int lane = threadIdx.x & 31;
    float4 v = *((const float4*)(h + (size_t)row * H) + lane);
    float s  = v.x + v.y + v.z + v.w;
    float s2 = v.x * v.x + v.y * v.y + v.z * v.z + v.w * v.w;
#pragma unroll
    for (int off = 16; off > 0; off >>= 1) {
        s  += __shfl_xor(s, off, 32);
        s2 += __shfl_xor(s2, off, 32);
    }
    const float mu  = s * (1.f / H);
    const float var = s2 * (1.f / H) - mu * mu;
    const float inv = rsqrtf(var + 1e-5f);
    float4 xv = *((const float4*)(x + (size_t)row * H) + lane);
    float4 g  = *((const float4*)gamma + lane);
    float4 bb = *((const float4*)beta + lane);
    float4 o;
    o.x = (v.x - mu) * inv * g.x + bb.x + xv.x;
    o.y = (v.y - mu) * inv * g.y + bb.y + xv.y;
    o.z = (v.z - mu) * inv * g.z + bb.z + xv.z;
    o.w = (v.w - mu) * inv * g.w + bb.w + xv.w;
    *((float4*)(out + (size_t)row * H) + lane) = o;
}

// ---------------------------------------------------------------------------
// Host launcher
// ---------------------------------------------------------------------------
extern "C" void kernel_launch(void* const* d_in, const int* in_sizes, int n_in,
                              void* d_out, int out_size, void* d_ws, size_t ws_size,
                              hipStream_t stream) {
    const float* x     = (const float*)d_in[0];
    const int*   eidx  = (const int*)d_in[1];   // [2, E] int32
    const float* W     = (const float*)d_in[2]; // [L, 128, 128]
    const float* w_ih  = (const float*)d_in[3]; // [384, 128]
    const float* w_hh  = (const float*)d_in[4]; // [384, 128]
    const float* b_ih  = (const float*)d_in[5];
    const float* b_hh  = (const float*)d_in[6];
    const float* gamma = (const float*)d_in[7];
    const float* beta  = (const float*)d_in[8];
    float*       out   = (float*)d_out;

    const int       N = in_sizes[0] / H;          // 100000
    const long long E = (long long)in_sizes[1] / 2;
    const int       L = in_sizes[2] / (H * H);    // 4

    const size_t NH = (size_t)N * H;
    float*  m    = (float*)d_ws;
    float*  agg  = m + NH;
    float*  hbuf = agg + NH;
    __bf16* Wp   = (__bf16*)(hbuf + NH);           // L * 8*4*32*16 bf16
    __bf16* wihP = Wp + (size_t)L * 8 * 4 * 32 * 16;
    __bf16* whhP = wihP + (size_t)24 * 4 * 32 * 16;

    const int rowBlocks  = (N + 15) / 16;
    const int zeroBlocks = (int)((NH / 4 + 255) / 256);
    const int edgeBlocks = (int)((E + 7) / 8);
    const int lnBlocks   = (N + 7) / 8;

    // One-shot weight packing (deterministic, cheap, L2-resident afterwards).
    {
        int totW = L * 8 * 4 * 32 * 16;
        gnn_pack_W<<<(totW + 255) / 256, 256, 0, stream>>>(W, Wp, L);
        int totG = 24 * 4 * 32 * 16;
        gnn_pack_wT<<<(totG + 255) / 256, 256, 0, stream>>>(w_ih, wihP);
        gnn_pack_wT<<<(totG + 255) / 256, 256, 0, stream>>>(w_hh, whhP);
    }

    const float* hin = x;
    for (int l = 0; l < L; ++l) {
        const v16bf* Wpl = (const v16bf*)(Wp + (size_t)l * 8 * 4 * 32 * 16);
        gnn_gemm_msg<<<rowBlocks, 256, 0, stream>>>(hin, Wpl, m, N);
        gnn_zero4<<<zeroBlocks, 256, 0, stream>>>((float4*)agg, (long long)NH / 4);
        gnn_scatter<<<edgeBlocks, 256, 0, stream>>>(m, eidx, eidx + E, agg, E);
        gnn_gru<<<rowBlocks, 256, 0, stream>>>(agg, hin, (const v16bf*)wihP,
                                               (const v16bf*)whhP, b_ih, b_hh, hbuf, N);
        hin = hbuf;   // in-place-safe for l >= 1 (blocks own disjoint row tiles)
    }
    gnn_layernorm<<<lnBlocks, 256, 0, stream>>>(hbuf, x, gamma, beta, out, N);
}
